// Random_SP_Block_Wrapper_8203387535736
// MI455X (gfx1250) — compile-verified, matrix operand order checked by reading, not run on hardware
//
#include <hip/hip_runtime.h>
#include <hip/hip_bf16.h>
#include <math.h>

typedef __bf16 bf16_t;
typedef __attribute__((ext_vector_type(16))) __bf16 v16bf;
typedef __attribute__((ext_vector_type(8)))  __bf16 v8bf;
typedef __attribute__((ext_vector_type(8)))  float  v8f;

#define B_     8
#define N_     257
#define D_     1024
#define H_     16
#define DH_    64
#define MREAL  (B_ * N_)      // 2056
#define MPAD   2080           // 65 * 32
#define QPAD   272            // 17 * 16 (query pad)
#define NPAD   288            // 9  * 32 (key pad)
#define NALPHA 64

// ---------------------------------------------------------------------------
// helpers
// ---------------------------------------------------------------------------
static __device__ inline v16bf load_ab(const bf16_t* p0, const bf16_t* p1) {
  // ISA A/B tile layout: lane holds 8 bf16 at K=klo and 8 bf16 at K=klo+16
  union { v16bf v; v8bf h[2]; } u;
  u.h[0] = *(const v8bf*)p0;
  u.h[1] = *(const v8bf*)p1;
  return u.v;
}
static __device__ inline v8f wmma_bf16(v16bf a, v16bf b, v8f c) {
  return __builtin_amdgcn_wmma_f32_16x16x32_bf16(false, a, false, b, (short)0, c,
                                                 false, false);
}
static __device__ inline v8f zero8() {
  v8f z = {0.f, 0.f, 0.f, 0.f, 0.f, 0.f, 0.f, 0.f};
  return z;
}
static __device__ inline void wait_async0() {
#if __has_builtin(__builtin_amdgcn_s_wait_asynccnt)
  __builtin_amdgcn_s_wait_asynccnt(0);
#else
  asm volatile("s_wait_asynccnt 0" ::: "memory");
#endif
}

// ---------------------------------------------------------------------------
// weight fp32 (K x N) -> bf16 transposed (N x K)   (B operand: lane=n, contig K)
// ---------------------------------------------------------------------------
__global__ void k_transpose_bf16(const float* __restrict__ W,
                                 bf16_t* __restrict__ Wt, int K, int Nout) {
  long long idx = (long long)blockIdx.x * blockDim.x + threadIdx.x;
  long long total = (long long)K * Nout;
  if (idx >= total) return;
  int k = (int)(idx % K);
  int n = (int)(idx / K);
  Wt[idx] = (bf16_t)W[(long long)k * Nout + n];
}

// ---------------------------------------------------------------------------
// LayerNorm -> bf16, rows >= MREAL zero-filled (pad for WMMA M tiles)
// ---------------------------------------------------------------------------
__global__ void k_layernorm(const float* __restrict__ x,
                            const float* __restrict__ g,
                            const float* __restrict__ b,
                            bf16_t* __restrict__ out) {
  int row = blockIdx.x;
  int t = threadIdx.x;
  __shared__ float s1[256], s2[256];
  if (row >= MREAL) {
    for (int i = t; i < D_; i += 256) out[(size_t)row * D_ + i] = (bf16_t)0.0f;
    return;
  }
  const float* xr = x + (size_t)row * D_;
  float a0 = 0.f, a1 = 0.f;
  for (int i = t; i < D_; i += 256) { float v = xr[i]; a0 += v; a1 += v * v; }
  s1[t] = a0; s2[t] = a1; __syncthreads();
  for (int s = 128; s > 0; s >>= 1) {
    if (t < s) { s1[t] += s1[t + s]; s2[t] += s2[t + s]; }
    __syncthreads();
  }
  float mean = s1[0] * (1.0f / D_);
  float var  = s2[0] * (1.0f / D_) - mean * mean;
  float rstd = rsqrtf(var + 1e-6f);
  for (int i = t; i < D_; i += 256)
    out[(size_t)row * D_ + i] = (bf16_t)((xr[i] - mean) * rstd * g[i] + b[i]);
}

// ---------------------------------------------------------------------------
// WMMA GEMM, software-pipelined K loop (next-step tiles prefetched into regs).
// C(M x Nout) = A_bf16(MPAD x K, lda) @ Bt_bf16(Nout x K) + bias
// EPI 0: store fp32          EPI 1: out = resid + ls * C (fp32)
// EPI 2: store bf16 gelu(C)  (exact gelu)
// wave tile 32(M) x 64(N); block = 4 waves -> 32 x 256; grid (MPAD/32, Nout/256)
// ---------------------------------------------------------------------------
template <int EPI>
__global__ void __launch_bounds__(128)
k_gemm(const bf16_t* __restrict__ A, int lda,
       const bf16_t* __restrict__ Bt,
       const float* __restrict__ bias, int K, int Nout,
       float* __restrict__ outF,
       const float* __restrict__ resid, const float* __restrict__ ls,
       bf16_t* __restrict__ outBf) {
  const int lane = threadIdx.x & 31;
  const int wave = threadIdx.x >> 5;
  const int half = lane >> 4;
  const int l16  = lane & 15;
  const int m0   = blockIdx.x * 32;
  const int n0   = blockIdx.y * 256 + wave * 64;
  const int klo  = half * 8;

  v8f acc[2][4];
#pragma unroll
  for (int i = 0; i < 2; i++)
#pragma unroll
    for (int j = 0; j < 4; j++) acc[i][j] = zero8();

  const bf16_t* a0p = A + (size_t)(m0 + l16) * lda + klo;
  const bf16_t* a1p = A + (size_t)(m0 + 16 + l16) * lda + klo;
  const bf16_t* b0p = Bt + (size_t)(n0 + 0 * 16 + l16) * K + klo;
  const bf16_t* b1p = Bt + (size_t)(n0 + 1 * 16 + l16) * K + klo;
  const bf16_t* b2p = Bt + (size_t)(n0 + 2 * 16 + l16) * K + klo;
  const bf16_t* b3p = Bt + (size_t)(n0 + 3 * 16 + l16) * K + klo;

  // prologue: load step 0 tiles
  v16bf av0 = load_ab(a0p, a0p + 16);
  v16bf av1 = load_ab(a1p, a1p + 16);
  v16bf bv0 = load_ab(b0p, b0p + 16);
  v16bf bv1 = load_ab(b1p, b1p + 16);
  v16bf bv2 = load_ab(b2p, b2p + 16);
  v16bf bv3 = load_ab(b3p, b3p + 16);

  const int steps = K >> 5;
  for (int s = 0; s < steps - 1; ++s) {
    const int kn = (s + 1) << 5;
    // L2->L0 prefetch one step further ahead (global_prefetch_b8)
    __builtin_prefetch((const void*)(a0p + kn + 32), 0, 1);
    __builtin_prefetch((const void*)(a1p + kn + 32), 0, 1);
    __builtin_prefetch((const void*)(b0p + kn + 32), 0, 1);
    __builtin_prefetch((const void*)(b1p + kn + 32), 0, 1);
    __builtin_prefetch((const void*)(b2p + kn + 32), 0, 1);
    __builtin_prefetch((const void*)(b3p + kn + 32), 0, 1);
    // issue next-step loads before consuming current tiles
    v16bf av0n = load_ab(a0p + kn, a0p + kn + 16);
    v16bf av1n = load_ab(a1p + kn, a1p + kn + 16);
    v16bf bv0n = load_ab(b0p + kn, b0p + kn + 16);
    v16bf bv1n = load_ab(b1p + kn, b1p + kn + 16);
    v16bf bv2n = load_ab(b2p + kn, b2p + kn + 16);
    v16bf bv3n = load_ab(b3p + kn, b3p + kn + 16);

    acc[0][0] = wmma_bf16(av0, bv0, acc[0][0]);
    acc[1][0] = wmma_bf16(av1, bv0, acc[1][0]);
    acc[0][1] = wmma_bf16(av0, bv1, acc[0][1]);
    acc[1][1] = wmma_bf16(av1, bv1, acc[1][1]);
    acc[0][2] = wmma_bf16(av0, bv2, acc[0][2]);
    acc[1][2] = wmma_bf16(av1, bv2, acc[1][2]);
    acc[0][3] = wmma_bf16(av0, bv3, acc[0][3]);
    acc[1][3] = wmma_bf16(av1, bv3, acc[1][3]);

    av0 = av0n; av1 = av1n;
    bv0 = bv0n; bv1 = bv1n; bv2 = bv2n; bv3 = bv3n;
  }
  // epilogue step
  acc[0][0] = wmma_bf16(av0, bv0, acc[0][0]);
  acc[1][0] = wmma_bf16(av1, bv0, acc[1][0]);
  acc[0][1] = wmma_bf16(av0, bv1, acc[0][1]);
  acc[1][1] = wmma_bf16(av1, bv1, acc[1][1]);
  acc[0][2] = wmma_bf16(av0, bv2, acc[0][2]);
  acc[1][2] = wmma_bf16(av1, bv2, acc[1][2]);
  acc[0][3] = wmma_bf16(av0, bv3, acc[0][3]);
  acc[1][3] = wmma_bf16(av1, bv3, acc[1][3]);

#pragma unroll
  for (int i = 0; i < 2; i++) {
#pragma unroll
    for (int j = 0; j < 4; j++) {
#pragma unroll
      for (int r = 0; r < 8; r++) {
        int row = m0 + i * 16 + r + half * 8;   // C layout: M = r + 8*(lane>=16)
        int col = n0 + j * 16 + l16;            //           N = lane & 15
        if (row < MREAL) {
          float v = acc[i][j][r] + bias[col];
          if (EPI == 0) {
            outF[(size_t)row * Nout + col] = v;
          } else if (EPI == 1) {
            outF[(size_t)row * Nout + col] =
                resid[(size_t)row * Nout + col] + ls[col] * v;
          } else {
            float gl = 0.5f * v * (1.0f + erff(v * 0.70710678118654752f));
            outBf[(size_t)row * Nout + col] = (bf16_t)gl;
          }
        }
      }
    }
  }
}

// ---------------------------------------------------------------------------
// qkv fp32 (MREAL x 3072) -> per-head bf16:
//  qb (BH, QPAD, 64) pre-scaled by dh^-0.5 ; kb (BH, NPAD, 64) ; vT (BH, 64, NPAD)
// ---------------------------------------------------------------------------
__global__ void k_qkv_prep(const float* __restrict__ qkv,
                           bf16_t* __restrict__ qb, bf16_t* __restrict__ kb,
                           bf16_t* __restrict__ vT) {
  int idx = blockIdx.x * 256 + threadIdx.x;  // total = 128*NPAD*64
  int d  = idx & 63;
  int n  = (idx >> 6) % NPAD;
  int bh = idx / (NPAD * 64);
  int b = bh >> 4, h = bh & 15;
  float qv = 0.f, kv = 0.f, vv = 0.f;
  if (n < N_) {
    size_t base = ((size_t)(b * N_ + n)) * (3 * D_) + h * DH_ + d;
    qv = qkv[base];
    kv = qkv[base + D_];
    vv = qkv[base + 2 * D_];
  }
  if (n < QPAD) qb[((size_t)bh * QPAD + n) * DH_ + d] = (bf16_t)(qv * 0.125f);
  kb[((size_t)bh * NPAD + n) * DH_ + d] = (bf16_t)kv;
  vT[((size_t)bh * DH_ + d) * NPAD + n] = (bf16_t)vv;
}

// ---------------------------------------------------------------------------
// Attention: one wave per (b,h, 16-query tile).
// stage1: logits 16 x 288 via WMMA (18 tiles, K=dh in 2 steps)
// softmax in registers (shfl_xor over 16-lane halves), bf16 probs -> LDS
// stage2: out 16 x 64 via WMMA (A from LDS, B from vT), scaled by 1/rowsum
// ---------------------------------------------------------------------------
__global__ void __launch_bounds__(32)
k_attention(const bf16_t* __restrict__ qb, const bf16_t* __restrict__ kb,
            const bf16_t* __restrict__ vT, bf16_t* __restrict__ outb) {
  const int bh = blockIdx.x;
  const int mt = blockIdx.y;
  const int b = bh >> 4, h = bh & 15;
  const int lane = threadIdx.x & 31;
  const int half = lane >> 4;
  const int l16  = lane & 15;
  const int klo  = half * 8;

  __shared__ alignas(16) bf16_t lp[16 * NPAD];

  const bf16_t* qrow = qb + ((size_t)bh * QPAD + mt * 16 + l16) * DH_ + klo;
  v16bf aq0 = load_ab(qrow, qrow + 16);        // d 0..31
  v16bf aq1 = load_ab(qrow + 32, qrow + 48);   // d 32..63

  v8f s[18];
#pragma unroll
  for (int t = 0; t < 18; t++) s[t] = zero8();

#pragma unroll
  for (int t = 0; t < 18; t++) {
    const bf16_t* krow = kb + ((size_t)bh * NPAD + t * 16 + l16) * DH_ + klo;
    v16bf bk0 = load_ab(krow, krow + 16);
    v16bf bk1 = load_ab(krow + 32, krow + 48);
    s[t] = wmma_bf16(aq0, bk0, s[t]);
    s[t] = wmma_bf16(aq1, bk1, s[t]);
  }

  float rsum[8];
#pragma unroll
  for (int r = 0; r < 8; r++) {
    float m = -INFINITY;
#pragma unroll
    for (int t = 0; t < 18; t++) {
      int col = t * 16 + l16;
      float v = (col < N_) ? s[t][r] : -INFINITY;
      m = fmaxf(m, v);
    }
    m = fmaxf(m, __shfl_xor(m, 1));
    m = fmaxf(m, __shfl_xor(m, 2));
    m = fmaxf(m, __shfl_xor(m, 4));
    m = fmaxf(m, __shfl_xor(m, 8));
    float sum = 0.f;
#pragma unroll
    for (int t = 0; t < 18; t++) {
      int col = t * 16 + l16;
      float p = (col < N_) ? __expf(s[t][r] - m) : 0.f;
      sum += p;
      lp[(r + half * 8) * NPAD + col] = (bf16_t)p;
    }
    sum += __shfl_xor(sum, 1);
    sum += __shfl_xor(sum, 2);
    sum += __shfl_xor(sum, 4);
    sum += __shfl_xor(sum, 8);
    rsum[r] = sum;
  }
  __syncthreads();

  v8f o[4];
#pragma unroll
  for (int j = 0; j < 4; j++) o[j] = zero8();
#pragma unroll
  for (int ks = 0; ks < 9; ks++) {
    int k0 = ks * 32;
    const bf16_t* ap = lp + l16 * NPAD + k0 + klo;
    v16bf pa = load_ab(ap, ap + 16);           // DS_LOAD_B128 A tiles from LDS
#pragma unroll
    for (int j = 0; j < 4; j++) {
      const bf16_t* vp =
          vT + ((size_t)bh * DH_ + j * 16 + l16) * NPAD + k0 + klo;
      v16bf vb = load_ab(vp, vp + 16);
      o[j] = wmma_bf16(pa, vb, o[j]);
    }
  }

#pragma unroll
  for (int j = 0; j < 4; j++) {
#pragma unroll
    for (int r = 0; r < 8; r++) {
      int qn = mt * 16 + r + half * 8;
      if (qn < N_) {
        float v = o[j][r] / rsum[r];
        outb[((size_t)(b * N_ + qn)) * D_ + h * DH_ + j * 16 + l16] = (bf16_t)v;
      }
    }
  }
}

// ---------------------------------------------------------------------------
// pad-row zeroing via CDNA5 async LDS->global store:
// build a zero tile in LDS, then global_store_async_from_lds_b128
// (VADDR = 64-bit global address, VSRC = LDS byte address per lane).
// count must be a multiple of 8 bf16 (16 bytes per lane).
// ---------------------------------------------------------------------------
__global__ void k_zero_bf16(bf16_t* p, long long count) {
  __shared__ alignas(16) bf16_t zbuf[256 * 8];   // only LDS object -> offset 0
  const int t = threadIdx.x;
#pragma unroll
  for (int j = 0; j < 8; j++) zbuf[t * 8 + j] = (bf16_t)0.0f;
  long long i0 = ((long long)blockIdx.x * 256 + t) * 8;
  if (i0 + 8 <= count) {
    unsigned long long ga = (unsigned long long)(p + i0);
    unsigned int lds_addr = (unsigned int)(t * 16);
    asm volatile("s_wait_dscnt 0" ::: "memory");  // LDS writes visible to async unit
    asm volatile("global_store_async_from_lds_b128 %0, %1, off"
                 :: "v"(ga), "v"(lds_addr)
                 : "memory");
    wait_async0();
  }
}

// ---------------------------------------------------------------------------
// greedy diversified selection + gather. One 256-thread block per batch.
// running-max over cluster similarities == reference's masked max over slots<k
// ---------------------------------------------------------------------------
__global__ void k_select(const float* __restrict__ x2,
                         const float* __restrict__ scores,
                         float* __restrict__ out) {
  const int b = blockIdx.x;
  const int t = threadIdx.x;
  __shared__ float curr[256];
  __shared__ float invn[256];
  __shared__ float red[256];
  __shared__ int   redi[256];
  __shared__ float cj[D_];
  __shared__ int   sel[NALPHA];

  const float* xb = x2 + (size_t)b * N_ * D_;
  {
    const float* r = xb + (size_t)(t + 1) * D_;
    float ss = 0.f;
    for (int d = 0; d < D_; d++) ss += r[d] * r[d];
    invn[t] = rsqrtf(ss);
  }
  curr[t] = scores[b * 256 + t];
  float my_msim = -INFINITY;
  __syncthreads();

  red[t] = curr[t]; redi[t] = t; __syncthreads();
  for (int s = 128; s > 0; s >>= 1) {
    if (t < s && red[t + s] > red[t]) { red[t] = red[t + s]; redi[t] = redi[t + s]; }
    __syncthreads();
  }
  if (t == 0) { sel[0] = redi[0]; curr[redi[0]] = -INFINITY; }
  __syncthreads();

  for (int k = 1; k < NALPHA; k++) {
    int j = sel[k - 1];
    float inj = invn[j];
    const float* rj = xb + (size_t)(j + 1) * D_;
    for (int d = t; d < D_; d += 256) cj[d] = rj[d] * inj;
    __syncthreads();
    const float* rp = xb + (size_t)(t + 1) * D_;
    float dot = 0.f;
    for (int d = 0; d < D_; d++) dot += rp[d] * cj[d];
    my_msim = fmaxf(my_msim, dot * invn[t]);
    red[t] = curr[t] - 0.2f * my_msim; redi[t] = t; __syncthreads();
    for (int s = 128; s > 0; s >>= 1) {
      if (t < s && red[t + s] > red[t]) { red[t] = red[t + s]; redi[t] = redi[t + s]; }
      __syncthreads();
    }
    if (t == 0) { sel[k] = redi[0]; curr[redi[0]] = -INFINITY; }
    __syncthreads();
  }

  if (t == 0) {  // ascending sort of selected patch indices
    for (int i = 1; i < NALPHA; i++) {
      int v = sel[i], j2 = i - 1;
      while (j2 >= 0 && sel[j2] > v) { sel[j2 + 1] = sel[j2]; j2--; }
      sel[j2 + 1] = v;
    }
  }
  __syncthreads();

  for (int i = 0; i <= NALPHA; i++) {
    int src = (i == 0) ? 0 : (sel[i - 1] + 1);
    for (int d = t; d < D_; d += 256)
      out[((size_t)b * (NALPHA + 1) + i) * D_ + d] = xb[(size_t)src * D_ + d];
  }
}

// ---------------------------------------------------------------------------
extern "C" void kernel_launch(void* const* d_in, const int* in_sizes, int n_in,
                              void* d_out, int out_size, void* d_ws,
                              size_t ws_size, hipStream_t stream) {
  (void)in_sizes; (void)n_in; (void)out_size; (void)ws_size;
  const float* x      = (const float*)d_in[0];
  const float* rsc    = (const float*)d_in[1];
  const float* n1g    = (const float*)d_in[2];
  const float* n1b    = (const float*)d_in[3];
  const float* qkv_w  = (const float*)d_in[4];
  const float* qkv_b  = (const float*)d_in[5];
  const float* proj_w = (const float*)d_in[6];
  const float* proj_b = (const float*)d_in[7];
  const float* ls1    = (const float*)d_in[8];
  const float* n2g    = (const float*)d_in[9];
  const float* n2b    = (const float*)d_in[10];
  const float* fc1_w  = (const float*)d_in[11];
  const float* fc1_b  = (const float*)d_in[12];
  const float* fc2_w  = (const float*)d_in[13];
  const float* fc2_b  = (const float*)d_in[14];
  const float* ls2    = (const float*)d_in[15];
  float* out = (float*)d_out;

  char* w = (char*)d_ws;
  size_t o = 0;
  auto alloc = [&](size_t bytes) -> char* {
    char* p = w + o;
    o = (o + bytes + 255) & ~(size_t)255;
    return p;
  };
  bf16_t* wqkv = (bf16_t*)alloc((size_t)3072 * 1024 * 2);
  bf16_t* wprj = (bf16_t*)alloc((size_t)1024 * 1024 * 2);
  bf16_t* wfc1 = (bf16_t*)alloc((size_t)4096 * 1024 * 2);
  bf16_t* wfc2 = (bf16_t*)alloc((size_t)1024 * 4096 * 2);
  bf16_t* h1   = (bf16_t*)alloc((size_t)MPAD * 1024 * 2);
  float*  qkvf = (float*) alloc((size_t)MREAL * 3072 * 4);
  bf16_t* qb   = (bf16_t*)alloc((size_t)128 * QPAD * 64 * 2);
  bf16_t* kb   = (bf16_t*)alloc((size_t)128 * NPAD * 64 * 2);
  bf16_t* vt   = (bf16_t*)alloc((size_t)128 * NPAD * 64 * 2);
  bf16_t* attn = (bf16_t*)alloc((size_t)MPAD * 1024 * 2);
  float*  x1   = (float*) alloc((size_t)MREAL * 1024 * 4);
  bf16_t* h2   = (bf16_t*)alloc((size_t)MPAD * 1024 * 2);
  bf16_t* gact = (bf16_t*)alloc((size_t)MPAD * 4096 * 2);
  float*  x2   = (float*) alloc((size_t)MREAL * 1024 * 4);

  // 0) weights -> bf16 transposed (N x K)
  k_transpose_bf16<<<(1024 * 3072) / 256, 256, 0, stream>>>(qkv_w, wqkv, 1024, 3072);
  k_transpose_bf16<<<(1024 * 1024) / 256, 256, 0, stream>>>(proj_w, wprj, 1024, 1024);
  k_transpose_bf16<<<(1024 * 4096) / 256, 256, 0, stream>>>(fc1_w, wfc1, 1024, 4096);
  k_transpose_bf16<<<(4096 * 1024) / 256, 256, 0, stream>>>(fc2_w, wfc2, 4096, 1024);

  // 1) LN1 -> bf16 (padded)
  k_layernorm<<<MPAD, 256, 0, stream>>>(x, n1g, n1b, h1);

  // 2) QKV GEMM (2080 x 1024) @ (3072 x 1024)^T -> fp32
  k_gemm<0><<<dim3(MPAD / 32, 3072 / 256), 128, 0, stream>>>(
      h1, 1024, wqkv, qkv_b, 1024, 3072, qkvf, nullptr, nullptr, nullptr);

  // 3) split/scale/pad/transpose q,k,v to bf16 head-major
  k_qkv_prep<<<(128 * NPAD * 64) / 256, 256, 0, stream>>>(qkvf, qb, kb, vt);

  // 4) attention (wave per b,h,query-tile)
  k_attention<<<dim3(128, 17), 32, 0, stream>>>(qb, kb, vt, attn);
  {
    long long cnt = (long long)(MPAD - MREAL) * 1024;   // 24576, /8 = 3072 lanes
    k_zero_bf16<<<(int)(cnt / 8 / 256), 256, 0, stream>>>(
        attn + (size_t)MREAL * 1024, cnt);
  }

  // 5) proj + residual: x1 = x + ls1 * (attn @ proj_w + proj_b)
  k_gemm<1><<<dim3(MPAD / 32, 1024 / 256), 128, 0, stream>>>(
      attn, 1024, wprj, proj_b, 1024, 1024, x1, x, ls1, nullptr);

  // 6) LN2 -> bf16 (padded)
  k_layernorm<<<MPAD, 256, 0, stream>>>(x1, n2g, n2b, h2);

  // 7) fc1 + exact gelu -> bf16
  k_gemm<2><<<dim3(MPAD / 32, 4096 / 256), 128, 0, stream>>>(
      h2, 1024, wfc1, fc1_b, 1024, 4096, nullptr, nullptr, nullptr, gact);
  {
    long long cnt = (long long)(MPAD - MREAL) * 4096;   // 98304, /8 = 12288 lanes
    k_zero_bf16<<<(int)(cnt / 8 / 256), 256, 0, stream>>>(
        gact + (size_t)MREAL * 4096, cnt);
  }

  // 8) fc2 + residual: x2 = x1 + ls2 * (g @ fc2_w + fc2_b)
  k_gemm<1><<<dim3(MPAD / 32, 1024 / 256), 128, 0, stream>>>(
      gact, 4096, wfc2, fc2_b, 4096, 1024, x2, x1, ls2, nullptr);

  // 9) diversified selection + gather -> (B, 65, 1024)
  k_select<<<B_, 256, 0, stream>>>(x2, rsc, out);
}
